// FavorPlusAttention_10093173145859
// MI455X (gfx1250) — matrix operand order repaired and checked by reading, not compiled, not run on hardware
//
#include <hip/hip_runtime.h>
#include <hip/hip_bf16.h>

// ---------------------------------------------------------------------------
// FAVOR+ attention for MI455X (gfx1250, wave32, WMMA bf16 16x16x32, f32 acc)
//
// Roofline: ~107 GFLOP total, ~1.5 GB HBM traffic with bf16 intermediates
// (~65 us at 23.3 TB/s) -> balanced; bf16 WMMA is the right precision path.
// GEMMs stage tiles into LDS via global_load_async_to_lds_b128 (ASYNCcnt),
// double-buffered so the next tile's DMA overlaps the current tile's WMMAs.
// ---------------------------------------------------------------------------

typedef __attribute__((ext_vector_type(16))) __bf16       v16bf;
typedef __attribute__((ext_vector_type(8)))  float        v8f;
typedef __attribute__((ext_vector_type(4)))  unsigned int uint4v;
typedef int v4i_gvec __attribute__((vector_size(16)));   // matches builtin param type

#define D_FEAT 256
#define D_POS  64
#define D_HID  320
#define HEADS  8
#define DHD    32
#define MFEAT  110          // real random-feature count
#define MPAD   128          // padded for WMMA K (zeros beyond 110)
#define NSEQ   325
#define NPAD   336          // 21 * 16
#define BL     384          // b*l
#define NTOK   (BL * NSEQ)  // 124800 (== 975 * 128, exact GEMM tiling)
#define NTOKP  (NTOK + 32)  // padded rows so edge-tile loads stay in bounds
#define NBH    (BL * HEADS) // 3072

#if __has_builtin(__builtin_amdgcn_global_load_async_to_lds_b128)
#define HAS_ASYNC_LDS 1
#else
#define HAS_ASYNC_LDS 0
#endif

static __device__ __forceinline__ void async_copy16(const void* g, void* l) {
#if HAS_ASYNC_LDS
    __builtin_amdgcn_global_load_async_to_lds_b128((v4i_gvec*)g, (v4i_gvec*)l, 0, 0);
#endif
}
static __device__ __forceinline__ void wait_async0() {
#if HAS_ASYNC_LDS
#if __has_builtin(__builtin_amdgcn_s_wait_asynccnt)
    __builtin_amdgcn_s_wait_asynccnt(0);
#else
    asm volatile("s_wait_asynccnt 0x0" ::: "memory");
#endif
#endif
}
static __device__ __forceinline__ void wait_async4() {
    // oldest 4 async ops done; 4 just-issued next-stage ops stay in flight
#if HAS_ASYNC_LDS
#if __has_builtin(__builtin_amdgcn_s_wait_asynccnt)
    __builtin_amdgcn_s_wait_asynccnt(4);
#else
    asm volatile("s_wait_asynccnt 0x4" ::: "memory");
#endif
#endif
}

static __device__ __forceinline__ float bf2f(__bf16 h) {
    unsigned short s = __builtin_bit_cast(unsigned short, h);
    unsigned int u = ((unsigned int)s) << 16;
    return __builtin_bit_cast(float, u);
}
static __device__ __forceinline__ __bf16 f2bf(float f) {
    unsigned int u = __builtin_bit_cast(unsigned int, f);
    unsigned int r = (u + 0x7fffu + ((u >> 16) & 1u)) >> 16;
    return __builtin_bit_cast(__bf16, (unsigned short)r);
}

union Frag16 {
    v16bf  v;
    uint4v q[2];
    __bf16 e[16];
};

static __device__ __forceinline__ v16bf load_frag(const __bf16* p0, const __bf16* p1) {
    Frag16 u;
    u.q[0] = *reinterpret_cast<const uint4v*>(p0);
    u.q[1] = *reinterpret_cast<const uint4v*>(p1);
    return u.v;
}

static __device__ __forceinline__ v8f v8f_zero() {
    v8f z;
#pragma unroll
    for (int i = 0; i < 8; ++i) z[i] = 0.0f;
    return z;
}

static __device__ __forceinline__ v8f wmma_bf16(v16bf a, v16bf b, v8f c) {
    // v_wmma_f32_16x16x32_bf16  D = A(16x32) * B(32x16) + C(16x16 f32)
    return __builtin_amdgcn_wmma_f32_16x16x32_bf16(false, a, false, b, (short)0, c, false, false);
}

// ------------------------- conversion / packing ----------------------------

__global__ void k_build_acat(const float* __restrict__ x, const float* __restrict__ ste,
                             __bf16* __restrict__ A) {
    long long i = (long long)blockIdx.x * blockDim.x + threadIdx.x;
    long long total = (long long)NTOK * D_HID;
    if (i >= total) return;
    int col = (int)(i % D_HID);
    long long t = i / D_HID;
    float v = (col < D_FEAT) ? x[t * D_FEAT + col] : ste[t * D_POS + (col - D_FEAT)];
    A[i] = f2bf(v);
}

__global__ void k_f32_to_bf16(const float* __restrict__ src, __bf16* __restrict__ dst, int n) {
    int i = blockIdx.x * blockDim.x + threadIdx.x;
    if (i < n) dst[i] = f2bf(src[i]);
}

__global__ void k_pack_wqkv(const float* __restrict__ wq, const float* __restrict__ wk,
                            const float* __restrict__ wv, __bf16* __restrict__ dst) {
    int i = blockIdx.x * blockDim.x + threadIdx.x;
    if (i >= 256 * 768) return;
    int k = i / 768, c = i % 768;
    float v = (c < 256) ? wq[k * 256 + c]
            : (c < 512) ? wk[k * 256 + (c - 256)]
                        : wv[k * 256 + (c - 512)];
    dst[i] = f2bf(v);
}

__global__ void k_pack_pt(const float* __restrict__ pm, __bf16* __restrict__ dst) {
    // PT[d][m] = proj_mat[m][d], m padded to 128 with zeros
    int i = blockIdx.x * blockDim.x + threadIdx.x;
    if (i >= DHD * MPAD) return;
    int d = i / MPAD, m = i % MPAD;
    dst[i] = f2bf((m < MFEAT) ? pm[m * DHD + d] : 0.0f);
}

// ------------------------------ tiled GEMM ---------------------------------
// C[M,N] = A[M,K](bf16,rowmajor) * B[K,N](bf16,rowmajor) (+bias)
// block = 256 threads (8 waves, 4x2), tile 128x128, BK=32.
// Each wave: 32x64 -> 2 A frags, 4 B frags, 8 WMMAs per K-step.
template<bool F32OUT, bool HASBIAS>
__global__ __launch_bounds__(256) void k_gemm(const __bf16* __restrict__ A,
                                              const __bf16* __restrict__ B,
                                              const float* __restrict__ bias,
                                              void* __restrict__ outp,
                                              int K, int N) {
    const int tid  = threadIdx.x;
    const int lane = tid & 31;
    const int wid  = tid >> 5;
    const int wm = wid >> 1, wn = wid & 1;
    const int l = lane & 15, hi = lane >> 4;

    const long long rowBase = (long long)blockIdx.x * 128;
    const int colBase = blockIdx.y * 128;

    const int arow = tid >> 1, acol = (tid & 1) * 16;
    const int brow = tid >> 3, bcol = (tid & 7) * 16;

    v8f acc[2][4];
#pragma unroll
    for (int i = 0; i < 2; ++i)
#pragma unroll
        for (int j = 0; j < 4; ++j) acc[i][j] = v8f_zero();

#if HAS_ASYNC_LDS
    // double-buffered async HBM -> LDS staging (ASYNCcnt pipelining)
    __shared__ __bf16 As[2][128][48];   // pitch 96B (16B-aligned rows)
    __shared__ __bf16 Bs[2][32][144];   // pitch 288B

    {   // prologue: stage k0 = 0 into buffer 0
        const __bf16* ga = A + (rowBase + arow) * K + acol;
        const __bf16* gb = B + (long long)brow * N + colBase + bcol;
        async_copy16(ga,     &As[0][arow][acol]);
        async_copy16(ga + 8, &As[0][arow][acol + 8]);
        async_copy16(gb,     &Bs[0][brow][bcol]);
        async_copy16(gb + 8, &Bs[0][brow][bcol + 8]);
    }
    int p = 0;
    for (int k0 = 0; k0 < K; k0 += 32, p ^= 1) {
        const bool hasNext = (k0 + 32 < K);
        __syncthreads();               // buf[p^1] free of last iteration's readers
        if (hasNext) {
            const __bf16* ga = A + (rowBase + arow) * K + (k0 + 32) + acol;
            const __bf16* gb = B + (long long)(k0 + 32 + brow) * N + colBase + bcol;
            async_copy16(ga,     &As[p ^ 1][arow][acol]);
            async_copy16(ga + 8, &As[p ^ 1][arow][acol + 8]);
            async_copy16(gb,     &Bs[p ^ 1][brow][bcol]);
            async_copy16(gb + 8, &Bs[p ^ 1][brow][bcol + 8]);
            wait_async4();             // stage-k data landed, stage-k+1 in flight
        } else {
            wait_async0();
        }
        __syncthreads();

        v16bf af[2], bfr[4];
#pragma unroll
        for (int i = 0; i < 2; ++i) {
            int r = wm * 32 + i * 16 + l;
            af[i] = load_frag(&As[p][r][hi ? 8 : 0], &As[p][r][hi ? 24 : 16]);
        }
#pragma unroll
        for (int j = 0; j < 4; ++j) {
            int c = wn * 64 + j * 16;
            Frag16 u;
#pragma unroll
            for (int s = 0; s < 16; ++s) u.e[s] = Bs[p][s < 8 ? 8 * hi + s : 8 + 8 * hi + s][c + l];
            bfr[j] = u.v;
        }
#pragma unroll
        for (int i = 0; i < 2; ++i)
#pragma unroll
            for (int j = 0; j < 4; ++j)
                acc[i][j] = wmma_bf16(af[i], bfr[j], acc[i][j]);
    }
#else
    // fallback: register-staged single buffer + prefetch
    __shared__ __bf16 As[128][48];
    __shared__ __bf16 Bs[32][144];

    for (int k0 = 0; k0 < K; k0 += 32) {
        const __bf16* ga = A + (rowBase + arow) * K + k0 + acol;
        const __bf16* gb = B + (long long)(k0 + brow) * N + colBase + bcol;
        uint4v a0 = reinterpret_cast<const uint4v*>(ga)[0];
        uint4v a1 = reinterpret_cast<const uint4v*>(ga)[1];
        uint4v b0 = reinterpret_cast<const uint4v*>(gb)[0];
        uint4v b1 = reinterpret_cast<const uint4v*>(gb)[1];
        if (k0 + 32 < K) {
            __builtin_prefetch(ga + 32, 0, 3);
            __builtin_prefetch(gb + (long long)32 * N, 0, 3);
        }
        __syncthreads();
        *reinterpret_cast<uint4v*>(&As[arow][acol])     = a0;
        *reinterpret_cast<uint4v*>(&As[arow][acol + 8]) = a1;
        *reinterpret_cast<uint4v*>(&Bs[brow][bcol])     = b0;
        *reinterpret_cast<uint4v*>(&Bs[brow][bcol + 8]) = b1;
        __syncthreads();

        v16bf af[2], bfr[4];
#pragma unroll
        for (int i = 0; i < 2; ++i) {
            int r = wm * 32 + i * 16 + l;
            af[i] = load_frag(&As[r][hi ? 8 : 0], &As[r][hi ? 24 : 16]);
        }
#pragma unroll
        for (int j = 0; j < 4; ++j) {
            int c = wn * 64 + j * 16;
            Frag16 u;
#pragma unroll
            for (int s = 0; s < 16; ++s) u.e[s] = Bs[s < 8 ? 8 * hi + s : 8 + 8 * hi + s][c + l];
            bfr[j] = u.v;
        }
#pragma unroll
        for (int i = 0; i < 2; ++i)
#pragma unroll
            for (int j = 0; j < 4; ++j)
                acc[i][j] = wmma_bf16(af[i], bfr[j], acc[i][j]);
    }
#endif

#pragma unroll
    for (int i = 0; i < 2; ++i)
#pragma unroll
        for (int j = 0; j < 4; ++j) {
            const int col = colBase + wn * 64 + j * 16 + l;
            const float badd = HASBIAS ? bias[col] : 0.0f;
#pragma unroll
            for (int r = 0; r < 8; ++r) {
                long long row = rowBase + wm * 32 + i * 16 + r + hi * 8;
                float v = acc[i][j][r] + badd;
                if constexpr (F32OUT) {
                    reinterpret_cast<float*>(outp)[row * N + col] = v;
                } else {
                    reinterpret_cast<__bf16*>(outp)[row * N + col] = f2bf(v);
                }
            }
        }
}

// ------------------------------ phi_q --------------------------------------
// One wave per (batch,head): dd = (q*dn) @ P^T, per-row max, exp -> phi_q.
__global__ __launch_bounds__(32) void k_phi_q(const __bf16* __restrict__ qkv,
                                              const __bf16* __restrict__ PT,
                                              __bf16* __restrict__ phiq) {
    const int bh = blockIdx.x;
    const int bl = bh >> 3, h = bh & 7;
    const int lane = threadIdx.x;
    const int l = lane & 15, hi = lane >> 4;
    __shared__ float sdiag[16];

    v16bf pf[8];
#pragma unroll
    for (int j = 0; j < 8; ++j) {
        Frag16 u;
        for (int s = 0; s < 16; ++s) u.e[s] = PT[(s < 8 ? 8 * hi + s : 8 + 8 * hi + s) * MPAD + j * 16 + l];
        pf[j] = u.v;
    }

    const float dn    = 0.42044820762685725f;   // 32^-0.25
    const float dn2h  = 0.5f * dn * dn;
    const float ratio = 0.09534625892455924f;   // 110^-0.5

    for (int tile = 0; tile < 21; ++tile) {
        int n0 = tile * 16;
        long long t = (long long)bl * NSEQ + n0 + l;
        const __bf16* qp = qkv + t * 768 + h * DHD;
        Frag16 a;
        a.q[0] = *reinterpret_cast<const uint4v*>(qp + (hi ? 8 : 0));
        a.q[1] = *reinterpret_cast<const uint4v*>(qp + (hi ? 24 : 16));
        float ss = 0.0f;
#pragma unroll
        for (int e = 0; e < 16; ++e) { float q = bf2f(a.e[e]); ss += q * q; }
        ss += __shfl_xor(ss, 16, 32);
        __syncthreads();
        if (hi == 0) sdiag[l] = dn2h * ss;
        __syncthreads();

        v8f dd[8];
#pragma unroll
        for (int j = 0; j < 8; ++j) dd[j] = wmma_bf16(a.v, pf[j], v8f_zero());

        float mx[8];
#pragma unroll
        for (int r = 0; r < 8; ++r) mx[r] = -3.0e38f;
#pragma unroll
        for (int j = 0; j < 8; ++j) {
            int m = l + j * 16;
            if (m < MFEAT)
#pragma unroll
                for (int r = 0; r < 8; ++r) mx[r] = fmaxf(mx[r], dn * dd[j][r]);
        }
#pragma unroll
        for (int r = 0; r < 8; ++r)
            for (int off = 1; off < 16; off <<= 1)
                mx[r] = fmaxf(mx[r], __shfl_xor(mx[r], off, 32));

#pragma unroll
        for (int j = 0; j < 8; ++j) {
            int m = l + j * 16;
#pragma unroll
            for (int r = 0; r < 8; ++r) {
                int rl = r + hi * 8;
                int n = n0 + rl;
                float v = 0.0f;
                if (m < MFEAT && n < NSEQ)
                    v = ratio * (__expf(dn * dd[j][r] - sdiag[rl] - mx[r]) + 1e-4f);
                phiq[((long long)bh * NPAD + n) * MPAD + m] = f2bf(v);
            }
        }
    }
}

// ------------------------------ phi_k + ctx --------------------------------
// 2 waves per (batch,head).  Pass1: global max + per-row diag.
// Pass2: phi_k tiles -> LDS (transposed), accumulate ksum[128] and
//        ctx[128x32] = phi_k^T @ v with WMMA.
__global__ __launch_bounds__(64) void k_phi_k_ctx(const __bf16* __restrict__ qkv,
                                                  const __bf16* __restrict__ PT,
                                                  __bf16* __restrict__ ctx,
                                                  float* __restrict__ ksum) {
    const int bh = blockIdx.x;
    const int bl = bh >> 3, h = bh & 7;
    const int tid = threadIdx.x;
    const int lane = tid & 31, wid = tid >> 5;
    const int l = lane & 15, hi = lane >> 4;

    __shared__ float  sdiag[352];
    __shared__ __bf16 phiT[MPAD][48];   // [m][n-in-chunk], pitch 96B
    __shared__ float  swmax[2];

    v16bf pf[8];
#pragma unroll
    for (int j = 0; j < 8; ++j) {
        Frag16 u;
        for (int s = 0; s < 16; ++s) u.e[s] = PT[(s < 8 ? 8 * hi + s : 8 + 8 * hi + s) * MPAD + j * 16 + l];
        pf[j] = u.v;
    }

    if (tid < 16) sdiag[336 + tid] = 0.0f;

    const float dn    = 0.42044820762685725f;
    const float dn2h  = 0.5f * dn * dn;
    const float ratio = 0.09534625892455924f;

    // ---- pass 1: diag + global max over valid (n,m) ----
    float gmax = -3.0e38f;
    for (int tile = wid; tile < 21; tile += 2) {
        int n0 = tile * 16;
        long long t = (long long)bl * NSEQ + n0 + l;
        const __bf16* kp = qkv + t * 768 + 256 + h * DHD;
        Frag16 a;
        a.q[0] = *reinterpret_cast<const uint4v*>(kp + (hi ? 8 : 0));
        a.q[1] = *reinterpret_cast<const uint4v*>(kp + (hi ? 24 : 16));
        float ss = 0.0f;
#pragma unroll
        for (int e = 0; e < 16; ++e) { float q = bf2f(a.e[e]); ss += q * q; }
        ss += __shfl_xor(ss, 16, 32);
        if (hi == 0) sdiag[n0 + l] = dn2h * ss;
#pragma unroll
        for (int j = 0; j < 8; ++j) {
            v8f dd = wmma_bf16(a.v, pf[j], v8f_zero());
            int m = l + j * 16;
            if (m < MFEAT) {
#pragma unroll
                for (int r = 0; r < 8; ++r) {
                    int n = n0 + r + hi * 8;
                    if (n < NSEQ) gmax = fmaxf(gmax, dn * dd[r]);
                }
            }
        }
    }
    for (int off = 1; off < 32; off <<= 1) gmax = fmaxf(gmax, __shfl_xor(gmax, off, 32));
    if (lane == 0) swmax[wid] = gmax;
    __syncthreads();
    const float MX = fmaxf(swmax[0], swmax[1]);

    // ---- pass 2 ----
    v8f cacc[4][2];
#pragma unroll
    for (int i = 0; i < 4; ++i)
#pragma unroll
        for (int d = 0; d < 2; ++d) cacc[i][d] = v8f_zero();
    float ka0 = 0.0f, ka1 = 0.0f;

    for (int chunk = 0; chunk < 11; ++chunk) {
        __syncthreads();                       // protect phiT from prior readers
        int n0 = chunk * 32 + wid * 16;
        long long t = (long long)bl * NSEQ + n0 + l;
        const __bf16* kp = qkv + t * 768 + 256 + h * DHD;
        Frag16 a;
        a.q[0] = *reinterpret_cast<const uint4v*>(kp + (hi ? 8 : 0));
        a.q[1] = *reinterpret_cast<const uint4v*>(kp + (hi ? 24 : 16));
#pragma unroll
        for (int j = 0; j < 8; ++j) {
            v8f dd = wmma_bf16(a.v, pf[j], v8f_zero());
            int m = l + j * 16;
#pragma unroll
            for (int r = 0; r < 8; ++r) {
                int rl = r + hi * 8;
                int n = n0 + rl;
                float v = 0.0f;
                if (m < MFEAT && n < NSEQ)
                    v = ratio * (__expf(dn * dd[r] - sdiag[n] - MX) + 1e-4f);
                phiT[m][wid * 16 + rl] = f2bf(v);
            }
        }
        __syncthreads();

        // ksum partials (thread t owns m = t and t+64)
        {
            const int m0 = tid, m1 = tid + 64;
#pragma unroll
            for (int n = 0; n < 32; ++n) {
                ka0 += bf2f(phiT[m0][n]);
                ka1 += bf2f(phiT[m1][n]);
            }
        }

        // ctx accumulate: wave wid owns m-tiles 4*wid .. 4*wid+3
        long long tv = (long long)bl * NSEQ + chunk * 32;
        const __bf16* vp = qkv + tv * 768 + 512 + h * DHD;
        v16bf bfr[2];
        for (int d = 0; d < 2; ++d) {
            Frag16 u;
            for (int s = 0; s < 16; ++s) u.e[s] = vp[(long long)(s < 8 ? 8 * hi + s : 8 + 8 * hi + s) * 768 + d * 16 + l];
            bfr[d] = u.v;
        }
#pragma unroll
        for (int i = 0; i < 4; ++i) {
            int mrow = (4 * wid + i) * 16 + l;
            v16bf af = load_frag(&phiT[mrow][hi ? 8 : 0], &phiT[mrow][hi ? 24 : 16]);
#pragma unroll
            for (int d = 0; d < 2; ++d)
                cacc[i][d] = wmma_bf16(af, bfr[d], cacc[i][d]);
        }
    }

#pragma unroll
    for (int i = 0; i < 4; ++i)
#pragma unroll
        for (int d = 0; d < 2; ++d)
#pragma unroll
            for (int r = 0; r < 8; ++r) {
                int m = (4 * wid + i) * 16 + r + hi * 8;
                int dcol = d * 16 + l;
                ctx[((long long)bh * MPAD + m) * DHD + dcol] = f2bf(cacc[i][d][r]);
            }
    ksum[(long long)bh * MPAD + tid]      = ka0;
    ksum[(long long)bh * MPAD + tid + 64] = ka1;
}

// ------------------------------ attn out -----------------------------------
// One wave per (batch,head, 16-row tile): out = (phi_q @ ctx) * d_inv.
__global__ __launch_bounds__(32) void k_attn_out(const __bf16* __restrict__ phiq,
                                                 const __bf16* __restrict__ ctx,
                                                 const float* __restrict__ ksum,
                                                 __bf16* __restrict__ attn) {
    const int bh = blockIdx.x;
    const int tile = blockIdx.y;
    const int bl = bh >> 3, h = bh & 7;
    const int lane = threadIdx.x;
    const int l = lane & 15, hi = lane >> 4;
    const int n0 = tile * 16;
    __shared__ float sdinv[16];

    if (hi == 0) {
        int n = n0 + l;
        float s = 0.0f;
        if (n < NSEQ) {
            const __bf16* pq = phiq + ((long long)bh * NPAD + n) * MPAD;
            const float* ks = ksum + (long long)bh * MPAD;
            for (int m = 0; m < MPAD; ++m) s += bf2f(pq[m]) * ks[m];
        }
        sdinv[l] = (n < NSEQ) ? 1.0f / s : 0.0f;
    }
    __syncthreads();

    v8f acc[2] = {v8f_zero(), v8f_zero()};
#pragma unroll
    for (int kc = 0; kc < 4; ++kc) {
        const __bf16* pq = phiq + ((long long)bh * NPAD + n0 + l) * MPAD + kc * 32;
        v16bf af = load_frag(pq + (hi ? 8 : 0), pq + (hi ? 24 : 16));
        const __bf16* cp = ctx + ((long long)bh * MPAD + kc * 32) * DHD;
        Frag16 u0, u1;
        for (int s = 0; s < 16; ++s) {
            int kk = s < 8 ? 8 * hi + s : 8 + 8 * hi + s;
            u0.e[s] = cp[kk * DHD + l];
            u1.e[s] = cp[kk * DHD + 16 + l];
        }
        v16bf b0 = u0.v;
        v16bf b1 = u1.v;
        acc[0] = wmma_bf16(af, b0, acc[0]);
        acc[1] = wmma_bf16(af, b1, acc[1]);
    }
#pragma unroll
    for (int d = 0; d < 2; ++d)
#pragma unroll
        for (int r = 0; r < 8; ++r) {
            int rl = r + hi * 8;
            int n = n0 + rl;
            if (n < NSEQ) {
                long long t = (long long)bl * NSEQ + n;
                attn[t * 256 + h * DHD + d * 16 + l] = f2bf(acc[d][r] * sdinv[rl]);
            }
        }
}

// ------------------------------ launcher -----------------------------------

extern "C" void kernel_launch(void* const* d_in, const int* in_sizes, int n_in,
                              void* d_out, int out_size, void* d_ws, size_t ws_size,
                              hipStream_t stream) {
    (void)in_sizes; (void)n_in; (void)out_size; (void)ws_size;

    const float* x        = (const float*)d_in[0];
    const float* ste      = (const float*)d_in[1];
    const float* proj_w   = (const float*)d_in[2];
    const float* proj_b   = (const float*)d_in[3];
    const float* wq       = (const float*)d_in[4];
    const float* wk       = (const float*)d_in[5];
    const float* wv       = (const float*)d_in[6];
    const float* wo       = (const float*)d_in[7];
    const float* wo_b     = (const float*)d_in[8];
    const float* proj_mat = (const float*)d_in[9];
    float* out = (float*)d_out;

    char* ws = (char*)d_ws;
    size_t off = 0;
    auto alloc = [&](size_t bytes) -> void* {
        void* p = ws + off;
        off = (off + bytes + 255) & ~(size_t)255;
        return p;
    };

    __bf16* acat  = (__bf16*)alloc((size_t)NTOK * D_HID * 2);     // ~80 MB
    __bf16* wprojB= (__bf16*)alloc((size_t)D_HID * D_FEAT * 2);
    __bf16* wqkvB = (__bf16*)alloc((size_t)256 * 768 * 2);
    __bf16* woB   = (__bf16*)alloc((size_t)256 * 256 * 2);
    __bf16* ptB   = (__bf16*)alloc((size_t)DHD * MPAD * 2);
    __bf16* hB    = (__bf16*)alloc((size_t)NTOK * D_FEAT * 2);    // ~64 MB
    __bf16* qkvB  = (__bf16*)alloc((size_t)NTOKP * 768 * 2);      // ~192 MB (padded rows)
    __bf16* phiqB = (__bf16*)alloc((size_t)NBH * NPAD * MPAD * 2);// ~264 MB
    __bf16* ctxB  = (__bf16*)alloc((size_t)NBH * MPAD * DHD * 2); // ~25 MB
    float*  ksumB = (float*) alloc((size_t)NBH * MPAD * 4);
    __bf16* attnB = (__bf16*)alloc((size_t)NTOK * 256 * 2);       // ~64 MB

    // 1) pack/convert
    {
        long long total = (long long)NTOK * D_HID;
        k_build_acat<<<(unsigned)((total + 255) / 256), 256, 0, stream>>>(x, ste, acat);
        k_f32_to_bf16<<<(D_HID * D_FEAT + 255) / 256, 256, 0, stream>>>(proj_w, wprojB, D_HID * D_FEAT);
        k_f32_to_bf16<<<(256 * 256 + 255) / 256, 256, 0, stream>>>(wo, woB, 256 * 256);
        k_pack_wqkv<<<(256 * 768 + 255) / 256, 256, 0, stream>>>(wq, wk, wv, wqkvB);
        k_pack_pt<<<(DHD * MPAD + 255) / 256, 256, 0, stream>>>(proj_mat, ptB);
    }
    // 2) h = [x|ste] @ proj_w + proj_b      (M=124800, K=320, N=256)
    k_gemm<false, true><<<dim3(NTOK / 128, D_FEAT / 128), 256, 0, stream>>>(
        acat, wprojB, proj_b, hB, D_HID, D_FEAT);
    // 3) qkv = h @ [wq|wk|wv]               (K=256, N=768)
    k_gemm<false, false><<<dim3(NTOK / 128, 768 / 128), 256, 0, stream>>>(
        hB, wqkvB, nullptr, qkvB, D_FEAT, 768);
    // 4) phi_q
    k_phi_q<<<NBH, 32, 0, stream>>>(qkvB, ptB, phiqB);
    // 5) phi_k -> ksum, ctx
    k_phi_k_ctx<<<NBH, 64, 0, stream>>>(qkvB, ptB, ctxB, ksumB);
    // 6) attn = (phi_q @ ctx) * d_inv, merged heads
    k_attn_out<<<dim3(NBH, 21), 32, 0, stream>>>(phiqB, ctxB, ksumB, attnB);
    // 7) out = attn @ wo + wo_b  -> f32 d_out
    k_gemm<true, true><<<dim3(NTOK / 128, D_FEAT / 128), 256, 0, stream>>>(
        attnB, woB, wo_b, out, D_FEAT, D_FEAT);
}
